// AttentionKVSplitted_51135880626369
// MI455X (gfx1250) — matrix operand order repaired and checked by reading, not compiled, not run on hardware
//
#include <hip/hip_runtime.h>
#include <hip/hip_bf16.h>
#include <stdint.h>

#define HEADS    8
#define DIM_HEAD 64
#define REPS     256
#define BATCH    8
#define SEQ_N    2048
#define SEQ_M    8192
#define DMODEL   512
#define TOPK     32

typedef __bf16 bf16_t;
typedef __attribute__((ext_vector_type(16))) __bf16 v16bf;
typedef __attribute__((ext_vector_type(8)))  float  v8f;

union FragAB { v16bf v; uint4 u[2]; };

__device__ __forceinline__ v8f wmma_bf16(v16bf a, v16bf b, v8f c) {
  return __builtin_amdgcn_wmma_f32_16x16x32_bf16(false, a, false, b, (short)0, c, false, false);
}

// CDNA5 async global->LDS copy (16 bytes), tracked by ASYNCcnt.
// LDS address operand = low 32 bits of the generic pointer (flat->LDS truncation).
__device__ __forceinline__ void async_copy16(const bf16_t* gsrc, bf16_t* lds_dst) {
  uint32_t loff = (uint32_t)(uintptr_t)lds_dst;
  asm volatile("global_load_async_to_lds_b128 %0, %1, off"
               :: "v"(loff), "v"((uint64_t)(uintptr_t)gsrc)
               : "memory");
}
__device__ __forceinline__ void wait_async0() {
  asm volatile("s_wait_asynccnt 0" ::: "memory");
}

// ---------------------------------------------------------------------------
// Kernel 1: exact f32 e0[b,:] = (x[b,0,:] @ Wq) @ We   (selection-critical)
// ---------------------------------------------------------------------------
__global__ __launch_bounds__(256)
void precompute_e0_kernel(const float* __restrict__ x, const float* __restrict__ Wq,
                          const float* __restrict__ We, float* __restrict__ e0) {
  const int b = blockIdx.x, t = threadIdx.x;
  __shared__ float q0[DMODEL];
  const float* xr = x + (size_t)b * SEQ_N * DMODEL;  // row n=0
  float a0 = 0.f, a1 = 0.f;
  for (int r = 0; r < DMODEL; ++r) {
    float xv = xr[r];
    a0 += xv * Wq[(size_t)r * DMODEL + t];
    a1 += xv * Wq[(size_t)r * DMODEL + t + 256];
  }
  q0[t] = a0; q0[t + 256] = a1;
  __syncthreads();
  float e = 0.f;
  for (int r = 0; r < DMODEL; ++r) e += q0[r] * We[(size_t)r * REPS + t];
  e0[b * REPS + t] = e;
}

// ---------------------------------------------------------------------------
// Kernel 2: dist2[b,m] = || context[b,m,:256] - e0[b,:] ||^2  (wave per m)
// ---------------------------------------------------------------------------
__global__ __launch_bounds__(256)
void dist_kernel(const float* __restrict__ context, const float* __restrict__ e0,
                 float* __restrict__ dist2) {
  const int lane = threadIdx.x & 31, w = threadIdx.x >> 5;
  const int m = blockIdx.x * 8 + w, b = blockIdx.y;
  const float* c = context + ((size_t)b * SEQ_M + m) * DMODEL;
  const float* e = e0 + b * REPS;
  float acc = 0.f;
  for (int i = 0; i < 8; ++i) {
    float d = c[lane + i * 32] - e[lane + i * 32];
    acc += d * d;
  }
  for (int off = 16; off; off >>= 1) acc += __shfl_xor(acc, off, 32);
  if (lane == 0) dist2[(size_t)b * SEQ_M + m] = acc;
}

// ---------------------------------------------------------------------------
// Kernel 3: top-32 smallest distances per batch (tie-break: lower index).
// ---------------------------------------------------------------------------
__global__ __launch_bounds__(256)
void topk_kernel(const float* __restrict__ dist2, int* __restrict__ idxout) {
  const int b = blockIdx.x, tid = threadIdx.x;
  __shared__ float sv[SEQ_M];
  __shared__ float rv[256];
  __shared__ int   ri[256];
  for (int i = tid; i < SEQ_M; i += 256) sv[i] = dist2[(size_t)b * SEQ_M + i];
  __syncthreads();
  for (int t = 0; t < TOPK; ++t) {
    float bv = 3.402823466e38f; int bi = 0x7fffffff;
    for (int i = tid; i < SEQ_M; i += 256) {
      float v = sv[i];
      if (v < bv || (v == bv && i < bi)) { bv = v; bi = i; }
    }
    rv[tid] = bv; ri[tid] = bi;
    __syncthreads();
    for (int s = 128; s; s >>= 1) {
      if (tid < s) {
        float ov = rv[tid + s]; int oi = ri[tid + s];
        if (ov < rv[tid] || (ov == rv[tid] && oi < ri[tid])) { rv[tid] = ov; ri[tid] = oi; }
      }
      __syncthreads();
    }
    if (tid == 0) { idxout[b * TOPK + t] = ri[0]; sv[ri[0]] = 3.402823466e38f; }
    __syncthreads();
  }
}

// ---------------------------------------------------------------------------
// Kernel 4: K/V projection of retrieved rows -> bf16 (b, h, j, d) layout.
// ---------------------------------------------------------------------------
__global__ __launch_bounds__(256)
void kv_kernel(const float* __restrict__ context, const int* __restrict__ idx,
               const float* __restrict__ Wk, const float* __restrict__ Wv,
               bf16_t* __restrict__ kbuf, bf16_t* __restrict__ vbuf) {
  const int j = blockIdx.x & (TOPK - 1), b = blockIdx.x / TOPK;
  const int t = threadIdx.x;
  __shared__ float row[DMODEL];
  const int ridx = idx[b * TOPK + j];
  const float* src = context + ((size_t)b * SEQ_M + ridx) * DMODEL;
  row[t] = src[t]; row[t + 256] = src[t + 256];
  __syncthreads();
  float k0 = 0.f, k1 = 0.f, v0 = 0.f, v1 = 0.f;
  for (int r = 0; r < REPS; ++r) {
    float lab = row[REPS + r], rep = row[r];
    k0 += lab * Wk[(size_t)r * DMODEL + t];
    k1 += lab * Wk[(size_t)r * DMODEL + t + 256];
    v0 += rep * Wv[(size_t)r * DMODEL + t];
    v1 += rep * Wv[(size_t)r * DMODEL + t + 256];
  }
  int c0 = t, c1 = t + 256;
  size_t o0 = ((size_t)(b * HEADS + (c0 >> 6)) * TOPK + j) * DIM_HEAD + (c0 & 63);
  size_t o1 = ((size_t)(b * HEADS + (c1 >> 6)) * TOPK + j) * DIM_HEAD + (c1 & 63);
  kbuf[o0] = (bf16_t)k0; kbuf[o1] = (bf16_t)k1;
  vbuf[o0] = (bf16_t)v0; vbuf[o1] = (bf16_t)v1;
}

// ---------------------------------------------------------------------------
// Kernel 5a: f32 -> bf16 bulk convert (n divisible by 1024).
// ---------------------------------------------------------------------------
__global__ __launch_bounds__(256)
void cvt_bf16_kernel(const float* __restrict__ in, bf16_t* __restrict__ out) {
  size_t i = ((size_t)blockIdx.x * 256 + threadIdx.x) * 4;
  float4 d = *(const float4*)(in + i);
  union { bf16_t h[4]; uint2 u; } p;
  p.h[0] = (bf16_t)d.x; p.h[1] = (bf16_t)d.y;
  p.h[2] = (bf16_t)d.z; p.h[3] = (bf16_t)d.w;
  *(uint2*)(out + i) = p.u;
}

// ---------------------------------------------------------------------------
// Kernel 5b: W[K][N] f32 -> WT[N][K] bf16 (tiled LDS transpose).
// ---------------------------------------------------------------------------
__global__ __launch_bounds__(256)
void transpose_w_kernel(const float* __restrict__ W, bf16_t* __restrict__ WT,
                        int Nn, int Kk) {
  __shared__ float tile[32][33];
  const int bx = blockIdx.x * 32;  // n base
  const int by = blockIdx.y * 32;  // k base
  const int tx = threadIdx.x & 31, ty = threadIdx.x >> 5;
  for (int i = 0; i < 4; ++i) {
    int k = by + ty + i * 8;
    tile[ty + i * 8][tx] = W[(size_t)k * Nn + bx + tx];
  }
  __syncthreads();
  for (int i = 0; i < 4; ++i) {
    int n = bx + ty + i * 8;
    WT[(size_t)n * Kk + by + tx] = (bf16_t)tile[tx][ty + i * 8];
  }
}

// ---------------------------------------------------------------------------
// Kernel 6: bf16 WMMA GEMM  C[M,N] = A[M,K] @ BT[N,K]^T (+bias)
// A bf16 row-major, BT bf16 row-major (pre-transposed weight).
// Block 64x64 (4 waves, 2x2 16x16 tiles each), K-step 32.
// A tile staged global->LDS with async DMA, double-buffered (ASYNCcnt).
// B fragments load straight from global (L2-resident weights).
// ---------------------------------------------------------------------------
template<bool OUTBF16>
__global__ __launch_bounds__(128)
void gemm_kernel(const bf16_t* __restrict__ A, const bf16_t* __restrict__ BT,
                 const float* __restrict__ bias, void* __restrict__ Cp,
                 int M, int N, int K) {
  __shared__ bf16_t ldsA[2][64 * 40];  // double-buffered [row][k], stride 40 halfs
  const int tid = threadIdx.x;
  const int lane = tid & 31, wave = tid >> 5;
  const int hl = lane & 15, hh = lane >> 4;
  const int wm = (wave & 1) * 32, wn = (wave >> 1) * 32;
  const int bm = blockIdx.y * 64, bn = blockIdx.x * 64;

  v8f acc[2][2] = {};

  // staging pattern: 64 rows x 32 halfs; 2x 16B async copies per thread
  const int srow = tid >> 1;            // 0..63
  const int sseg = (tid & 1) * 16;      // half offset 0 or 16

  // prologue: issue tile 0
  {
    const bf16_t* g = A + (size_t)(bm + srow) * K + sseg;
    bf16_t* l = &ldsA[0][srow * 40 + sseg];
    async_copy16(g, l);
    async_copy16(g + 8, l + 8);
  }

  const int nsteps = K / 32;
  for (int s = 0; s < nsteps; ++s) {
    wait_async0();        // my async copies for tile s done
    __syncthreads();      // everyone's copies visible
    const bf16_t* cur = ldsA[s & 1];
    if (s + 1 < nsteps) { // issue tile s+1 into the other buffer
      const bf16_t* g = A + (size_t)(bm + srow) * K + (s + 1) * 32 + sseg;
      bf16_t* l = &ldsA[(s + 1) & 1][srow * 40 + sseg];
      async_copy16(g, l);
      async_copy16(g + 8, l + 8);
    }
    const int k0 = s * 32;

    FragAB a[2], b[2];
    for (int mi = 0; mi < 2; ++mi) {
      const bf16_t* p0 = &cur[(wm + mi * 16 + hl) * 40 + hh * 8];
      a[mi].u[0] = *(const uint4*)p0;
      a[mi].u[1] = *(const uint4*)(p0 + 16);
    }
    for (int ni = 0; ni < 2; ++ni) {
      const bf16_t* p0 = BT + (size_t)(bn + wn + ni * 16 + hl) * K + k0 + hh * 8;
      b[ni].u[0] = *(const uint4*)p0;
      b[ni].u[1] = *(const uint4*)(p0 + 16);
    }
    for (int mi = 0; mi < 2; ++mi)
      for (int ni = 0; ni < 2; ++ni)
        acc[mi][ni] = wmma_bf16(a[mi].v, b[ni].v, acc[mi][ni]);
  }

  for (int mi = 0; mi < 2; ++mi)
    for (int ni = 0; ni < 2; ++ni) {
      int gcol = bn + wn + ni * 16 + hl;
      float bv = bias ? bias[gcol] : 0.f;
      for (int r = 0; r < 8; ++r) {
        int grow = bm + wm + mi * 16 + r + hh * 8;
        float v = acc[mi][ni][r] + bv;
        if (OUTBF16) ((bf16_t*)Cp)[(size_t)grow * N + gcol] = (bf16_t)v;
        else         ((float*)Cp)[(size_t)grow * N + gcol] = v;
      }
    }
}

// ---------------------------------------------------------------------------
// Kernel 7: fused attention. One wave per (b, head, 16-row tile of N).
// ---------------------------------------------------------------------------
__global__ __launch_bounds__(32)
void attn_kernel(const bf16_t* __restrict__ q, const bf16_t* __restrict__ kbuf,
                 const bf16_t* __restrict__ vbuf, bf16_t* __restrict__ aout) {
  const int lane = threadIdx.x;
  const int hl = lane & 15, hh = lane >> 4;
  const int n0 = blockIdx.x * 16;
  const int head = blockIdx.y;
  const int b = blockIdx.z;

  __shared__ bf16_t vT[64 * 40];   // [d][j] transposed V tile
  __shared__ bf16_t aP[16 * 40];   // attention probs [m][j]

  const bf16_t* kb = kbuf + (size_t)(b * HEADS + head) * TOPK * DIM_HEAD;
  const bf16_t* vb = vbuf + (size_t)(b * HEADS + head) * TOPK * DIM_HEAD;

  for (int i = lane; i < DIM_HEAD * TOPK; i += 32) {
    int d = i >> 5, j = i & 31;
    vT[d * 40 + j] = vb[j * DIM_HEAD + d];
  }

  // sim: A = q tile (16x64 -> two K=32 steps), B = k^T (contiguous kbuf rows)
  v8f s0 = {}, s1 = {};
  const bf16_t* qrow = q + ((size_t)(b * SEQ_N) + n0 + hl) * DMODEL + head * DIM_HEAD;
  for (int dstep = 0; dstep < 2; ++dstep) {
    int dbase = dstep * 32;
    FragAB a;
    a.u[0] = *(const uint4*)(qrow + dbase + hh * 8);
    a.u[1] = *(const uint4*)(qrow + dbase + 16 + hh * 8);
    FragAB b0, b1;
    const bf16_t* kr0 = kb + (size_t)hl * DIM_HEAD + dbase + hh * 8;
    const bf16_t* kr1 = kb + (size_t)(16 + hl) * DIM_HEAD + dbase + hh * 8;
    b0.u[0] = *(const uint4*)kr0; b0.u[1] = *(const uint4*)(kr0 + 16);
    b1.u[0] = *(const uint4*)kr1; b1.u[1] = *(const uint4*)(kr1 + 16);
    s0 = wmma_bf16(a.v, b0.v, s0);
    s1 = wmma_bf16(a.v, b1.v, s1);
  }

  // softmax over j (32): row m = r + 8*hh lives on the 16 lanes sharing hh.
  const float scale = 0.125f;  // 64^-0.5
  for (int r = 0; r < 8; ++r) {
    float x0 = s0[r] * scale, x1 = s1[r] * scale;
    float mx = fmaxf(x0, x1);
    for (int off = 8; off; off >>= 1) mx = fmaxf(mx, __shfl_xor(mx, off, 32));
    float p0 = __expf(x0 - mx), p1 = __expf(x1 - mx);
    float sm = p0 + p1;
    for (int off = 8; off; off >>= 1) sm += __shfl_xor(sm, off, 32);
    float inv = 1.f / sm;
    int row = r + hh * 8;
    aP[row * 40 + hl]      = (bf16_t)(p0 * inv);
    aP[row * 40 + 16 + hl] = (bf16_t)(p1 * inv);
  }
  __syncthreads();

  // out = attn (16x32) @ v (32x64): A from aP, B from vT, 4 column tiles.
  FragAB pa;
  { const bf16_t* p0 = &aP[hl * 40 + hh * 8];
    pa.u[0] = *(const uint4*)p0;
    pa.u[1] = *(const uint4*)(p0 + 16); }

  v8f o[4] = {};
  for (int dt = 0; dt < 4; ++dt) {
    const bf16_t* p0 = &vT[(dt * 16 + hl) * 40 + hh * 8];
    FragAB vb2;
    vb2.u[0] = *(const uint4*)p0;
    vb2.u[1] = *(const uint4*)(p0 + 16);
    o[dt] = wmma_bf16(pa.v, vb2.v, o[dt]);
  }
  for (int dt = 0; dt < 4; ++dt)
    for (int r = 0; r < 8; ++r) {
      int m = r + hh * 8;
      int col = head * DIM_HEAD + dt * 16 + hl;
      aout[((size_t)(b * SEQ_N) + n0 + m) * DMODEL + col] = (bf16_t)o[dt][r];
    }
}

// ---------------------------------------------------------------------------
// Host side
// ---------------------------------------------------------------------------
extern "C" void kernel_launch(void* const* d_in, const int* in_sizes, int n_in,
                              void* d_out, int out_size, void* d_ws, size_t ws_size,
                              hipStream_t stream) {
  (void)in_sizes; (void)n_in; (void)out_size; (void)ws_size;
  const float* x       = (const float*)d_in[0];
  const float* context = (const float*)d_in[1];
  const float* Wq      = (const float*)d_in[2];
  const float* Wk      = (const float*)d_in[3];
  const float* Wv      = (const float*)d_in[4];
  const float* We      = (const float*)d_in[5];
  const float* Wo      = (const float*)d_in[6];
  const float* bo      = (const float*)d_in[7];

  char* ws = (char*)d_ws;
  constexpr size_t OFF_E0   = 0;                       // 8 KB
  constexpr size_t OFF_DIST = OFF_E0   + 8192;         // 256 KB
  constexpr size_t OFF_IDX  = OFF_DIST + 262144;       // 1 KB
  constexpr size_t OFF_K    = OFF_IDX  + 1024;         // 256 KB
  constexpr size_t OFF_V    = OFF_K    + 262144;       // 256 KB
  constexpr size_t OFF_WQT  = OFF_V    + 262144;       // 512*512*2 = 512 KB
  constexpr size_t OFF_WOT  = OFF_WQT  + 524288;       // 512 KB
  constexpr size_t OFF_XB   = OFF_WOT  + 524288;       // 16 MB
  constexpr size_t OFF_Q    = OFF_XB   + 16777216;     // 16 MB
  constexpr size_t OFF_AOUT = OFF_Q    + 16777216;     // 16 MB

  float*  e0    = (float*)(ws + OFF_E0);
  float*  dist2 = (float*)(ws + OFF_DIST);
  int*    idxp  = (int*)(ws + OFF_IDX);
  bf16_t* kbuf  = (bf16_t*)(ws + OFF_K);
  bf16_t* vbuf  = (bf16_t*)(ws + OFF_V);
  bf16_t* WqT   = (bf16_t*)(ws + OFF_WQT);
  bf16_t* WoT   = (bf16_t*)(ws + OFF_WOT);
  bf16_t* xb    = (bf16_t*)(ws + OFF_XB);
  bf16_t* qbuf  = (bf16_t*)(ws + OFF_Q);
  bf16_t* aoutb = (bf16_t*)(ws + OFF_AOUT);

  const int Mrows = BATCH * SEQ_N;  // 16384

  // precision prep: bf16 activations + transposed bf16 weights
  cvt_bf16_kernel<<<(Mrows * DMODEL) / 1024, 256, 0, stream>>>(x, xb);
  transpose_w_kernel<<<dim3(DMODEL / 32, DMODEL / 32), 256, 0, stream>>>(Wq, WqT, DMODEL, DMODEL);
  transpose_w_kernel<<<dim3(DMODEL / 32, DMODEL / 32), 256, 0, stream>>>(Wo, WoT, DMODEL, DMODEL);

  // q = x @ Wq  (async-staged bf16 WMMA)
  gemm_kernel<true><<<dim3(DMODEL / 64, Mrows / 64), 128, 0, stream>>>(
      xb, WqT, (const float*)nullptr, (void*)qbuf, Mrows, DMODEL, DMODEL);

  // exact f32 selection path
  precompute_e0_kernel<<<BATCH, 256, 0, stream>>>(x, Wq, We, e0);
  dist_kernel<<<dim3(SEQ_M / 8, BATCH), 256, 0, stream>>>(context, e0, dist2);
  topk_kernel<<<BATCH, 256, 0, stream>>>(dist2, idxp);
  kv_kernel<<<BATCH * TOPK, 256, 0, stream>>>(context, idxp, Wk, Wv, kbuf, vbuf);

  // fused attention
  attn_kernel<<<dim3(SEQ_N / 16, HEADS, BATCH), 32, 0, stream>>>(qbuf, kbuf, vbuf, aoutb);

  // out = attn_out @ Wo + bo  (f32 output)
  gemm_kernel<false><<<dim3(DMODEL / 64, Mrows / 64), 128, 0, stream>>>(
      aoutb, WoT, bo, d_out, Mrows, DMODEL, DMODEL);
}
